// EncoderBlock_2911987826714
// MI455X (gfx1250) — matrix operand order repaired and checked by reading, not compiled, not run on hardware
//
#include <hip/hip_runtime.h>
#include <hip/hip_bf16.h>

typedef __attribute__((ext_vector_type(16))) _Float16 v16h;
typedef __attribute__((ext_vector_type(8)))  float    v8f;

// ---------------------------------------------------------------------------
// WMMA helpers (CDNA5 16x16x32 f16 -> f32)
// ---------------------------------------------------------------------------
__device__ __forceinline__ v8f wmma_f16(v16h a, v16h b, v8f c) {
  return __builtin_amdgcn_wmma_f32_16x16x32_f16(
      /*neg_a=*/false, a, /*neg_b=*/false, b,
      /*c_mod=*/(short)0, c, /*reuse_a=*/false, /*reuse_b=*/false);
}

// A-matrix fragment (16x32, M = lane&15, K pattern per ISA 7.12.2):
//   VGPR j (pair): k = 2j + (j>=4 ? 8 : 0) + 8*grp
// Pairs are contiguous -> compiler merges these into two ds_load_b128.
__device__ __forceinline__ v16h frag_a(const _Float16* __restrict__ p, int grp) {
  union { v16h v; unsigned u[8]; } f;
#pragma unroll
  for (int j = 0; j < 8; ++j) {
    int k = 2 * j + ((j >= 4) ? 8 : 0) + 8 * grp;
    f.u[j] = *(const unsigned*)(p + k);
  }
  return f.v;
}

// B-matrix fragment (32x16, N = lane&15, staged [N][K]):
//   VGPR j (pair): k = 2j + (j>=4 ? 8 : 0) + 8*grp  -> merges into two ds_load_b128.
__device__ __forceinline__ v16h frag_b(const _Float16* __restrict__ p, int grp) {
  union { v16h v; unsigned u[8]; } f;
#pragma unroll
  for (int j = 0; j < 8; ++j) {
    int k = 2 * j + ((j >= 4) ? 8 : 0) + 8 * grp;
    f.u[j] = *(const unsigned*)(p + k);
  }
  return f.v;
}

// ---------------------------------------------------------------------------
// Weight convert + transpose: w[K][N] (f32) -> wt[N][K] (f16)
// ---------------------------------------------------------------------------
__global__ void cvt_transpose_h(const float* __restrict__ w,
                                _Float16* __restrict__ wt, int K, int N) {
  int idx = blockIdx.x * 256 + threadIdx.x;
  if (idx >= K * N) return;
  int n = idx / K, k = idx - n * K;
  wt[(size_t)n * K + k] = (_Float16)w[(size_t)k * N + n];
}

// ---------------------------------------------------------------------------
// LayerNorm (ddof=1, eps added to std) -> f16 row
// ---------------------------------------------------------------------------
__global__ __launch_bounds__(256, 1)
void ln_to_h(const float* __restrict__ x, const float* __restrict__ alpha,
             const float* __restrict__ beta, _Float16* __restrict__ out, int D) {
  const int row = blockIdx.x;
  const float* xr = x + (size_t)row * D;
  float s = 0.f, s2 = 0.f;
  for (int i = threadIdx.x; i < D; i += 256) { float v = xr[i]; s += v; s2 += v * v; }
#pragma unroll
  for (int m = 16; m >= 1; m >>= 1) { s += __shfl_xor(s, m); s2 += __shfl_xor(s2, m); }
  __shared__ float rs[8], rs2[8];
  int w = threadIdx.x >> 5;
  if ((threadIdx.x & 31) == 0) { rs[w] = s; rs2[w] = s2; }
  __syncthreads();
  s = 0.f; s2 = 0.f;
#pragma unroll
  for (int i = 0; i < 8; ++i) { s += rs[i]; s2 += rs2[i]; }
  float mean = s / (float)D;
  float var  = (s2 - (float)D * mean * mean) / ((float)D - 1.f);
  float inv  = alpha[0] / (sqrtf(fmaxf(var, 0.f)) + 1e-6f);
  float b0   = beta[0];
  _Float16* orow = out + (size_t)row * D;
  for (int i = threadIdx.x; i < D; i += 256)
    orow[i] = (_Float16)((xr[i] - mean) * inv + b0);
}

// ---------------------------------------------------------------------------
// Generic WMMA GEMM: C[M][N] = A[M][K](f16,row-major) * Bt[N][K](f16)
// Block tile 128x128x64, 256 threads = 8 waves (4x2), wave tile 32x64,
// 16 WMMAs / wave / k-iter. Double-buffered LDS + register prefetch:
// one barrier per k-iteration, global latency overlapped with WMMA.
// Epilogue MODE:
//  1: Q proj  -> qh[b][h][s][d]  (scaled by 1/8)
//  2: K proj  -> kh[b][h][s][d]
//  3: V proj  -> vth[b][h][d][s] (transposed per head)
//  4: O proj  -> outF = resid + val              (fp32)
//  5: FFN1    -> outH = relu(val + bias[col])    (f16)
//  6: FFN2    -> outF = resid + val + bias[col]  (fp32)
// ---------------------------------------------------------------------------
#define GBM 128
#define GBN 128
#define GBK 64
#define GSTR 72   // LDS row stride in halves (144B = 9*16B -> float4-aligned)

template <int MODE>
__global__ __launch_bounds__(256, 1)
void gemm_wmma(const _Float16* __restrict__ A, const _Float16* __restrict__ Bt,
               int M, int N, int K,
               float* __restrict__ outF, const float* __restrict__ resid,
               _Float16* __restrict__ outH, const float* __restrict__ bias,
               int S) {
  __shared__ __align__(16) _Float16 As[2][GBM * GSTR];
  __shared__ __align__(16) _Float16 Bs[2][GBN * GSTR];
  const int t = threadIdx.x, lane = t & 31, wave = t >> 5;
  const int grp = lane >> 4, lr = lane & 15;
  const int wm = wave >> 1, wn = wave & 1;
  const int m0 = blockIdx.y * GBM, n0 = blockIdx.x * GBN;

  v8f acc[2][4];
  const v8f vzero = {0.f, 0.f, 0.f, 0.f, 0.f, 0.f, 0.f, 0.f};
#pragma unroll
  for (int mt = 0; mt < 2; ++mt)
#pragma unroll
    for (int nt = 0; nt < 4; ++nt) acc[mt][nt] = vzero;

  // preload k-tile 0 into buffer 0 (128x64 halves, 4 float4 per thread)
#pragma unroll
  for (int c = 0; c < 4; ++c) {
    int lin = (t + c * 256) * 8;
    int r = lin >> 6, cc = lin & 63;
    *(float4*)&As[0][r * GSTR + cc] = *(const float4*)(A + (size_t)(m0 + r) * K + cc);
    *(float4*)&Bs[0][r * GSTR + cc] = *(const float4*)(Bt + (size_t)(n0 + r) * K + cc);
  }
  __syncthreads();

  const int nk = K >> 6;
  for (int kt = 0; kt < nk; ++kt) {
    const int cur = kt & 1;
    const bool more = (kt + 1) < nk;
    float4 pa[4], pb[4];
    if (more) {
      const int k0 = (kt + 1) << 6;
#pragma unroll
      for (int c = 0; c < 4; ++c) {
        int lin = (t + c * 256) * 8;
        int r = lin >> 6, cc = lin & 63;
        pa[c] = *(const float4*)(A + (size_t)(m0 + r) * K + k0 + cc);
        pb[c] = *(const float4*)(Bt + (size_t)(n0 + r) * K + k0 + cc);
      }
    }

    const _Float16* Ab = As[cur];
    const _Float16* Bb = Bs[cur];
    v16h af[2][2];
#pragma unroll
    for (int mt = 0; mt < 2; ++mt)
#pragma unroll
      for (int kk = 0; kk < 2; ++kk)
        af[mt][kk] = frag_a(&Ab[(wm * 32 + mt * 16 + lr) * GSTR + kk * 32], grp);
#pragma unroll
    for (int nt = 0; nt < 4; ++nt) {
      v16h b0 = frag_b(&Bb[(wn * 64 + nt * 16 + lr) * GSTR + 0], grp);
      v16h b1 = frag_b(&Bb[(wn * 64 + nt * 16 + lr) * GSTR + 32], grp);
#pragma unroll
      for (int mt = 0; mt < 2; ++mt) {
        acc[mt][nt] = wmma_f16(af[mt][0], b0, acc[mt][nt]);
        acc[mt][nt] = wmma_f16(af[mt][1], b1, acc[mt][nt]);
      }
    }

    if (more) {
      const int nxt = cur ^ 1;
#pragma unroll
      for (int c = 0; c < 4; ++c) {
        int lin = (t + c * 256) * 8;
        int r = lin >> 6, cc = lin & 63;
        *(float4*)&As[nxt][r * GSTR + cc] = pa[c];
        *(float4*)&Bs[nxt][r * GSTR + cc] = pb[c];
      }
    }
    __syncthreads();
  }

  // epilogue
#pragma unroll
  for (int mt = 0; mt < 2; ++mt)
#pragma unroll
    for (int nt = 0; nt < 4; ++nt)
#pragma unroll
      for (int r = 0; r < 8; ++r) {
        int row = m0 + wm * 32 + mt * 16 + r + 8 * grp;
        int col = n0 + wn * 64 + nt * 16 + lr;
        float v = acc[mt][nt][r];
        if constexpr (MODE == 1 || MODE == 2 || MODE == 3) {
          int b = (row >= S) ? 1 : 0;  // B == 2
          int s = row - b * S;
          int h = col >> 6, d = col & 63;
          if constexpr (MODE == 1) {
            v *= 0.125f;  // 1/sqrt(dk) folded into Q
            outH[(((size_t)(b * 12 + h) * S) + s) * 64 + d] = (_Float16)v;
          } else if constexpr (MODE == 2) {
            outH[(((size_t)(b * 12 + h) * S) + s) * 64 + d] = (_Float16)v;
          } else {
            outH[(((size_t)(b * 12 + h) * 64) + d) * S + s] = (_Float16)v;
          }
        } else if constexpr (MODE == 4) {
          size_t o = (size_t)row * N + col;
          outF[o] = resid[o] + v;
        } else if constexpr (MODE == 5) {
          v += bias[col];
          v = fmaxf(v, 0.f);
          outH[(size_t)row * N + col] = (_Float16)v;
        } else if constexpr (MODE == 6) {
          size_t o = (size_t)row * N + col;
          outF[o] = resid[o] + v + bias[col];
        }
      }
}

// ---------------------------------------------------------------------------
// Flash attention: per (b,h), 64 query rows per block, 4 waves x 16 q-rows.
// qh/kh: [B][H][S][64] f16 (q pre-scaled), vth: [B][H][64][S] f16.
// Double-buffered K/V tiles with register prefetch (mask words prefetched a
// tile ahead too), one barrier per kv-tile. Online softmax; mask quirk:
// masked scores SET to 1e-9 before softmax (branchless select).
// Output oh: [B*S][768] f16 (col = h*64 + d).
// ---------------------------------------------------------------------------
#define FSTR 72   // LDS row stride in halves (144B, float4-aligned)

__global__ __launch_bounds__(128, 1)
void flash_attn(const _Float16* __restrict__ qh, const _Float16* __restrict__ kh,
                const _Float16* __restrict__ vth, const int* __restrict__ mask,
                _Float16* __restrict__ oh, int S) {
  __shared__ __align__(16) _Float16 Kt[2][64 * FSTR];
  __shared__ __align__(16) _Float16 Vt[2][64 * FSTR];
  __shared__ __align__(16) _Float16 Pl[4 * 16 * FSTR];

  const int b = blockIdx.z, h = blockIdx.y;
  const int q0 = blockIdx.x * 64;
  const int t = threadIdx.x, wave = t >> 5, lane = t & 31;
  const int grp = lane >> 4, lr = lane & 15;

  const _Float16* qbase = qh + ((size_t)(b * 12 + h) * S) * 64;
  const _Float16* kbase = kh + ((size_t)(b * 12 + h) * S) * 64;
  const _Float16* vbase = vth + ((size_t)(b * 12 + h) * 64) * S;
  const int* mbase = mask + (size_t)b * S;

  // Q fragments for this wave's 16 rows (K = 64 -> two 32-wide fragments)
  const int qrow = q0 + wave * 16 + lr;
  v16h qf0 = frag_a(qbase + (size_t)qrow * 64 + 0, grp);
  v16h qf1 = frag_a(qbase + (size_t)qrow * 64 + 32, grp);

  const v8f vzero = {0.f, 0.f, 0.f, 0.f, 0.f, 0.f, 0.f, 0.f};
  v8f oacc[4];
#pragma unroll
  for (int nt = 0; nt < 4; ++nt) oacc[nt] = vzero;
  float mrow[8], lrow[8];
#pragma unroll
  for (int r = 0; r < 8; ++r) { mrow[r] = -1e30f; lrow[r] = 0.f; }

  _Float16* pw = Pl + wave * 16 * FSTR;

  // preload kv-tile 0 into buffer 0 (each tile 64x64 halves, 4 float4/thread)
#pragma unroll
  for (int c = 0; c < 4; ++c) {
    int lin = (t + c * 128) * 8;
    int r = lin >> 6, cc = lin & 63;
    *(float4*)&Kt[0][r * FSTR + cc] = *(const float4*)(kbase + (size_t)r * 64 + cc);
    *(float4*)&Vt[0][r * FSTR + cc] = *(const float4*)(vbase + (size_t)r * S + cc);
  }
  // preload tile-0 mask words (one column word per nt)
  int mcur[4];
#pragma unroll
  for (int nt = 0; nt < 4; ++nt) mcur[nt] = mbase[nt * 16 + lr];
  __syncthreads();

  const int ntiles = S >> 6;
  for (int it = 0; it < ntiles; ++it) {
    const int cur = it & 1;
    const bool more = (it + 1) < ntiles;
    float4 pk[4], pv[4];
    int mnxt[4];
    if (more) {
      const int nkv = (it + 1) << 6;
#pragma unroll
      for (int c = 0; c < 4; ++c) {
        int lin = (t + c * 128) * 8;
        int r = lin >> 6, cc = lin & 63;
        pk[c] = *(const float4*)(kbase + (size_t)(nkv + r) * 64 + cc);
        pv[c] = *(const float4*)(vbase + (size_t)r * S + nkv + cc);
      }
#pragma unroll
      for (int nt = 0; nt < 4; ++nt) mnxt[nt] = mbase[nkv + nt * 16 + lr];
    }

    // scores: S16x64 = Q(16x64) @ K^T ; Bt staging = K rows [kv][d]
    const _Float16* Kb = Kt[cur];
    const _Float16* Vb = Vt[cur];
    v8f sacc[4];
#pragma unroll
    for (int nt = 0; nt < 4; ++nt) {
      sacc[nt] = vzero;
      v16h kf0 = frag_b(&Kb[(nt * 16 + lr) * FSTR + 0], grp);
      v16h kf1 = frag_b(&Kb[(nt * 16 + lr) * FSTR + 32], grp);
      sacc[nt] = wmma_f16(qf0, kf0, sacc[nt]);
      sacc[nt] = wmma_f16(qf1, kf1, sacc[nt]);
    }

    // mask quirk (branchless): score = 1e-9 where mask == 0
#pragma unroll
    for (int nt = 0; nt < 4; ++nt)
#pragma unroll
      for (int r = 0; r < 8; ++r)
        sacc[nt][r] = (mcur[nt] == 0) ? 1e-9f : sacc[nt][r];

    // online softmax: per-row max/sum across the 16 lanes of this half
    float sc[8];
#pragma unroll
    for (int r = 0; r < 8; ++r) {
      float mx = -1e30f;
#pragma unroll
      for (int nt = 0; nt < 4; ++nt) mx = fmaxf(mx, sacc[nt][r]);
#pragma unroll
      for (int m = 8; m >= 1; m >>= 1) mx = fmaxf(mx, __shfl_xor(mx, m));
      float nm = fmaxf(mrow[r], mx);
      sc[r] = __expf(mrow[r] - nm);
      mrow[r] = nm;
    }
#pragma unroll
    for (int r = 0; r < 8; ++r) {
      float rs = 0.f;
#pragma unroll
      for (int nt = 0; nt < 4; ++nt) {
        float p = __expf(sacc[nt][r] - mrow[r]);
        sacc[nt][r] = p;
        rs += p;
      }
#pragma unroll
      for (int m = 8; m >= 1; m >>= 1) rs += __shfl_xor(rs, m);
      lrow[r] = lrow[r] * sc[r] + rs;
#pragma unroll
      for (int nt = 0; nt < 4; ++nt) oacc[nt][r] *= sc[r];
    }

    // P: C-layout -> A-layout via per-wave LDS round-trip (wave-local order)
#pragma unroll
    for (int nt = 0; nt < 4; ++nt)
#pragma unroll
      for (int r = 0; r < 8; ++r)
        pw[(r + 8 * grp) * FSTR + nt * 16 + lr] = (_Float16)sacc[nt][r];
    asm volatile("s_wait_dscnt 0" ::: "memory");

    v16h pf0 = frag_a(pw + lr * FSTR + 0, grp);
    v16h pf1 = frag_a(pw + lr * FSTR + 32, grp);
#pragma unroll
    for (int nt = 0; nt < 4; ++nt) {
      v16h vf0 = frag_b(&Vb[(nt * 16 + lr) * FSTR + 0], grp);
      v16h vf1 = frag_b(&Vb[(nt * 16 + lr) * FSTR + 32], grp);
      oacc[nt] = wmma_f16(pf0, vf0, oacc[nt]);
      oacc[nt] = wmma_f16(pf1, vf1, oacc[nt]);
    }

    if (more) {
      const int nxt = cur ^ 1;
#pragma unroll
      for (int c = 0; c < 4; ++c) {
        int lin = (t + c * 128) * 8;
        int r = lin >> 6, cc = lin & 63;
        *(float4*)&Kt[nxt][r * FSTR + cc] = pk[c];
        *(float4*)&Vt[nxt][r * FSTR + cc] = pv[c];
      }
#pragma unroll
      for (int nt = 0; nt < 4; ++nt) mcur[nt] = mnxt[nt];
    }
    __syncthreads();
  }

  // write O / l  -> oh[b*S+s][h*64+d]
#pragma unroll
  for (int nt = 0; nt < 4; ++nt)
#pragma unroll
    for (int r = 0; r < 8; ++r) {
      int s = q0 + wave * 16 + r + 8 * grp;
      int d = nt * 16 + lr;
      float v = oacc[nt][r] / lrow[r];
      oh[((size_t)(b * S + s)) * 768 + h * 64 + d] = (_Float16)v;
    }
}

// ---------------------------------------------------------------------------
// Launcher
// ---------------------------------------------------------------------------
extern "C" void kernel_launch(void* const* d_in, const int* in_sizes, int n_in,
                              void* d_out, int out_size, void* d_ws, size_t ws_size,
                              hipStream_t stream) {
  const int B = 2, S = 2048, D = 768, F = 3072, H = 12;
  const int M = B * S;

  const float* x    = (const float*)d_in[0];
  const int*   mask = (const int*)d_in[1];
  const float* wq   = (const float*)d_in[2];
  const float* wk   = (const float*)d_in[3];
  const float* wv   = (const float*)d_in[4];
  const float* wo   = (const float*)d_in[5];
  const float* w1   = (const float*)d_in[6];
  const float* b1   = (const float*)d_in[7];
  const float* w2   = (const float*)d_in[8];
  const float* b2   = (const float*)d_in[9];
  const float* ln1a = (const float*)d_in[10];
  const float* ln1b = (const float*)d_in[11];
  const float* ln2a = (const float*)d_in[12];
  const float* ln2b = (const float*)d_in[13];
  float* outF = (float*)d_out;

  // workspace carve (f16 elements; every size is a multiple of 128 -> aligned)
  _Float16* base = (_Float16*)d_ws;
  size_t off = 0;
  auto carve = [&](size_t n) { _Float16* r = base + off; off += n; return r; };
  _Float16* wqT  = carve((size_t)D * D);
  _Float16* wkT  = carve((size_t)D * D);
  _Float16* wvT  = carve((size_t)D * D);
  _Float16* woT  = carve((size_t)D * D);
  _Float16* w1T  = carve((size_t)D * F);
  _Float16* w2T  = carve((size_t)D * F);
  _Float16* t_h  = carve((size_t)M * D);
  _Float16* q_h  = carve((size_t)M * D);
  _Float16* k_h  = carve((size_t)M * D);
  _Float16* vT_h = carve((size_t)M * D);
  _Float16* o_h  = carve((size_t)M * D);
  _Float16* h_h  = carve((size_t)M * F);

  // 1) weight convert + transpose to [N][K] f16
  {
    int tdd = D * D, tdf = D * F;
    cvt_transpose_h<<<(tdd + 255) / 256, 256, 0, stream>>>(wq, wqT, D, D);
    cvt_transpose_h<<<(tdd + 255) / 256, 256, 0, stream>>>(wk, wkT, D, D);
    cvt_transpose_h<<<(tdd + 255) / 256, 256, 0, stream>>>(wv, wvT, D, D);
    cvt_transpose_h<<<(tdd + 255) / 256, 256, 0, stream>>>(wo, woT, D, D);
    cvt_transpose_h<<<(tdf + 255) / 256, 256, 0, stream>>>(w1, w1T, D, F);
    cvt_transpose_h<<<(tdf + 255) / 256, 256, 0, stream>>>(w2, w2T, F, D);
  }

  // 2) LN1
  ln_to_h<<<M, 256, 0, stream>>>(x, ln1a, ln1b, t_h, D);

  // 3) QKV projections (Q pre-scaled by 1/8; V stored transposed per head)
  dim3 gdd(D / GBN, M / GBM);
  gemm_wmma<1><<<gdd, 256, 0, stream>>>(t_h, wqT, M, D, D, nullptr, nullptr, q_h, nullptr, S);
  gemm_wmma<2><<<gdd, 256, 0, stream>>>(t_h, wkT, M, D, D, nullptr, nullptr, k_h, nullptr, S);
  gemm_wmma<3><<<gdd, 256, 0, stream>>>(t_h, wvT, M, D, D, nullptr, nullptr, vT_h, nullptr, S);

  // 4) flash attention
  flash_attn<<<dim3(S / 64, H, B), 128, 0, stream>>>(q_h, k_h, vT_h, mask, o_h, S);

  // 5) output projection + residual 1 (fp32 x1 -> d_out)
  gemm_wmma<4><<<gdd, 256, 0, stream>>>(o_h, woT, M, D, D, outF, x, nullptr, nullptr, S);

  // 6) LN2 (reuse t_h)
  ln_to_h<<<M, 256, 0, stream>>>(outF, ln2a, ln2b, t_h, D);

  // 7) FFN
  dim3 gdf(F / GBN, M / GBM);
  gemm_wmma<5><<<gdf, 256, 0, stream>>>(t_h, w1T, M, F, D, nullptr, nullptr, h_h, b1, S);
  gemm_wmma<6><<<gdd, 256, 0, stream>>>(h_h, w2T, M, D, F, outF, outF, nullptr, b2, S);
}
